// GRUPushUpCounter_88519275971017
// MI455X (gfx1250) — matrix operand order repaired and checked by reading, not compiled, not run on hardware
//
#include <hip/hip_runtime.h>
#include <hip/hip_fp16.h>

typedef __attribute__((ext_vector_type(16))) _Float16 v16h;
typedef __attribute__((ext_vector_type(8)))  float    v8f;

#define BATCH 512
#define SEQT  2048
#define INF   6
#define HID   32

// gfx1250 has V_TANH_F32; use it if the builtin exists (guarded, compile-safe).
#if __has_builtin(__builtin_amdgcn_tanhf)
__device__ __forceinline__ float ftanh_(float x) { return __builtin_amdgcn_tanhf(x); }
#elif __has_builtin(__builtin_amdgcn_tanh_f32)
__device__ __forceinline__ float ftanh_(float x) { return __builtin_amdgcn_tanh_f32(x); }
#else
__device__ __forceinline__ float ftanh_(float x) { return 2.f / (1.f + __expf(-2.f * x)) - 1.f; }
#endif
__device__ __forceinline__ float fsig(float x) { return 0.5f * ftanh_(0.5f * x) + 0.5f; }

#define WMMA_F16(A, B, C) \
  __builtin_amdgcn_wmma_f32_16x16x32_f16(false, (A), false, (B), (short)0, (C), false, false)

union AOp { v16h v; uint32_t u[8]; _Float16 h[16]; };

// h0 is stored in WMMA-A-operand swizzled layout:
//   dword index = ((tile*SEQT + t)*2 + dir)*256 + lane*8 + i,  i in [0,8)
// where lane's 8 dwords are exactly its v16h A-operand image for that 16x32 tile.

// ---------------------------------------------------------------------------
// Layer 0 scan: x[B,T,6] -> h0 (swizzled f16). 12 WMMAs / step
// (input GEMM zero-padded K=6->32 + recurrent K=32). 64 waves total.
// ---------------------------------------------------------------------------
__global__ __launch_bounds__(32) void gru_layer0(
    const float* __restrict__ x,
    const float* __restrict__ WihF, const float* __restrict__ WhhF,
    const float* __restrict__ bihF, const float* __restrict__ bhhF,
    const float* __restrict__ WihB, const float* __restrict__ WhhB,
    const float* __restrict__ bihB, const float* __restrict__ bhhB,
    _Float16* __restrict__ h0)
{
  const int lane = threadIdx.x;
  const int nloc = lane & 15;     // A row / output column within 16-wide tile
  const int hi   = lane >> 4;     // K-half select in A/B striping
  const int dir  = blockIdx.x & 1;
  const int tile = blockIdx.x >> 1;
  const int b0   = tile * 16;

  const float* Wih = dir ? WihB : WihF;
  const float* Whh = dir ? WhhB : WhhF;
  const float* bih = dir ? bihB : bihF;
  const float* bhh = dir ? bhhB : bhhF;

  // Loop-invariant per-lane weights/biases. Tile t: gate g=t>>1, cols (t&1)*16..+15.
  v16h bwh[6], bwx[6];
  float bc[4], bin[2], bhn[2];
#pragma unroll
  for (int t = 0; t < 6; ++t) {
    const int row = (t >> 1) * 32 + (t & 1) * 16 + nloc;   // gate row in [0,96)
    const float bi_ = bih[row], bh_ = bhh[row];
    if (t < 4) bc[t] = bi_ + bh_;                 // r,z: biases combine
    else { bin[t - 4] = bi_; bhn[t - 4] = bh_; }  // n: keep split (PyTorch form)
#pragma unroll
    for (int e = 0; e < 16; ++e) {
      const int k = hi * 16 + e;                  // B element: B[k][n] = W[row][k]
      bwh[t][e] = (_Float16)Whh[row * HID + k];
      bwx[t][e] = (k < INF) ? (_Float16)Wih[row * INF + k] : (_Float16)0.f;
    }
  }

  __shared__ _Float16 lds_h[16 * HID];      // hidden state f16, [m][k]
  float hreg[16];                           // idx c*8+r <-> (m=r+8*hi, j=nloc+16*c)
#pragma unroll
  for (int i = 0; i < 16; ++i) hreg[i] = 0.f;

  AOp Ah;                                   // current-state A operand (h_{t-1})
#pragma unroll
  for (int i = 0; i < 8; ++i) Ah.u[i] = 0u; // h_{-1} = 0

  const int tstep = dir ? -1 : 1;
  const int tt0   = dir ? (SEQT - 1) : 0;
  const float xmask = hi ? 0.f : 1.f;       // hi lanes carry K>=16 -> zero

  const float* xp = x + ((size_t)(b0 + nloc) * SEQT + tt0) * INF;
  const int xadv  = tstep * INF;
  uint32_t* h0u   = (uint32_t*)h0;
  size_t sbase    = (((size_t)tile * SEQT + tt0) * 2 + dir) * 256 + (size_t)lane * 8;
  const long sadv = (long)tstep * 512;

  float xv[INF];
#pragma unroll
  for (int i = 0; i < INF; ++i) xv[i] = xp[i];

  for (int ts = 0; ts < SEQT; ++ts) {
    // A_x operand (16x32, zero-padded past K=6); lane owns x row nloc.
    AOp Ax;
#pragma unroll
    for (int i = 0; i < 8; ++i) Ax.u[i] = 0u;
#pragma unroll
    for (int e = 0; e < INF; ++e) Ax.h[e] = (_Float16)(xv[e] * xmask);

    // Prefetch next x row; overlaps WMMA + gate work below.
    xp += xadv;
    if (ts + 1 < SEQT) {
#pragma unroll
      for (int i = 0; i < INF; ++i) xv[i] = xp[i];
    }

    const v8f zero = {};
    v8f a0 = WMMA_F16(Ax.v, bwx[0], zero);
    v8f a1 = WMMA_F16(Ax.v, bwx[1], zero);
    v8f a2 = WMMA_F16(Ax.v, bwx[2], zero);
    v8f a3 = WMMA_F16(Ax.v, bwx[3], zero);
    a0 = WMMA_F16(Ah.v, bwh[0], a0);
    a1 = WMMA_F16(Ah.v, bwh[1], a1);
    a2 = WMMA_F16(Ah.v, bwh[2], a2);
    a3 = WMMA_F16(Ah.v, bwh[3], a3);
    v8f p4 = WMMA_F16(Ax.v, bwx[4], zero);
    v8f p5 = WMMA_F16(Ax.v, bwx[5], zero);
    v8f g4 = WMMA_F16(Ah.v, bwh[4], zero);
    v8f g5 = WMMA_F16(Ah.v, bwh[5], zero);

#pragma unroll
    for (int r = 0; r < 8; ++r) {
      const int m = r + 8 * hi;
      {
        const float rg = fsig(a0[r] + bc[0]);
        const float zg = fsig(a2[r] + bc[2]);
        const float ng = ftanh_(p4[r] + bin[0] + rg * (g4[r] + bhn[0]));
        const float hn = ng + zg * (hreg[r] - ng);
        hreg[r] = hn;
        lds_h[m * HID + nloc] = (_Float16)hn;
      }
      {
        const float rg = fsig(a1[r] + bc[1]);
        const float zg = fsig(a3[r] + bc[3]);
        const float ng = ftanh_(p5[r] + bin[1] + rg * (g5[r] + bhn[1]));
        const float hn = ng + zg * (hreg[8 + r] - ng);
        hreg[8 + r] = hn;
        lds_h[m * HID + 16 + nloc] = (_Float16)hn;
      }
    }
    __syncthreads();

    // Rebuild A operand for h_t once: feeds next step's WMMA AND the h0 store.
    {
      const uint32_t* hrow = (const uint32_t*)&lds_h[nloc * HID];
#pragma unroll
      for (int i = 0; i < 4; ++i) { Ah.u[i] = hrow[hi * 8 + i]; Ah.u[4 + i] = hrow[hi * 8 + 4 + i]; }
    }
    __syncthreads();

    // Store the lane's A-operand image: 2x b128, fully coalesced.
    uint32_t* sp = h0u + sbase;
    *(uint4*)(sp)     = make_uint4(Ah.u[0], Ah.u[1], Ah.u[2], Ah.u[3]);
    *(uint4*)(sp + 4) = make_uint4(Ah.u[4], Ah.u[5], Ah.u[6], Ah.u[7]);
    sbase += sadv;
  }
}

// ---------------------------------------------------------------------------
// Layer 1 scan: h0 (swizzled f16) -> pooled[B,64] f32 (mean over T of [f1|b1]).
// 18 WMMAs / step; A0/A1 load directly from global in operand layout
// (no LDS re-layout), prefetched one step ahead (h0 is L2-resident).
// ---------------------------------------------------------------------------
__global__ __launch_bounds__(32) void gru_layer1(
    const _Float16* __restrict__ h0,
    const float* __restrict__ WihF, const float* __restrict__ WhhF,
    const float* __restrict__ bihF, const float* __restrict__ bhhF,
    const float* __restrict__ WihB, const float* __restrict__ WhhB,
    const float* __restrict__ bihB, const float* __restrict__ bhhB,
    float* __restrict__ pooled)
{
  const int lane = threadIdx.x;
  const int nloc = lane & 15;
  const int hi   = lane >> 4;
  const int dir  = blockIdx.x & 1;
  const int tile = blockIdx.x >> 1;
  const int b0   = tile * 16;

  const float* Wih = dir ? WihB : WihF;
  const float* Whh = dir ? WhhB : WhhF;
  const float* bih = dir ? bihB : bihF;
  const float* bhh = dir ? bhhB : bhhF;

  v16h bwh[6], bwi0[6], bwi1[6];
  float bc[4], bin[2], bhn[2];
#pragma unroll
  for (int t = 0; t < 6; ++t) {
    const int row = (t >> 1) * 32 + (t & 1) * 16 + nloc;
    const float bi_ = bih[row], bh_ = bhh[row];
    if (t < 4) bc[t] = bi_ + bh_;
    else { bin[t - 4] = bi_; bhn[t - 4] = bh_; }
#pragma unroll
    for (int e = 0; e < 16; ++e) {
      bwh[t][e]  = (_Float16)Whh[row * HID + hi * 16 + e];
      bwi0[t][e] = (_Float16)Wih[row * 64 + hi * 16 + e];
      bwi1[t][e] = (_Float16)Wih[row * 64 + 32 + hi * 16 + e];
    }
  }

  __shared__ _Float16 lds_h[16 * HID];

  float hreg[16], pool[16];
#pragma unroll
  for (int i = 0; i < 16; ++i) { hreg[i] = 0.f; pool[i] = 0.f; }

  AOp Ah;
#pragma unroll
  for (int i = 0; i < 8; ++i) Ah.u[i] = 0u;

  const int tstep = dir ? -1 : 1;
  const int tt0   = dir ? (SEQT - 1) : 0;
  const uint32_t* h0u = (const uint32_t*)h0;
  size_t lbase    = ((size_t)tile * SEQT + tt0) * 512 + (size_t)lane * 8;
  const long ladv = (long)tstep * 512;

  // Preload first tile's operands (A0 = forward half, A1 = backward half).
  AOp A0, A1, N0, N1;
  {
    const uint32_t* p = h0u + lbase;
    uint4 a = *(const uint4*)(p);       uint4 b = *(const uint4*)(p + 4);
    uint4 c = *(const uint4*)(p + 256); uint4 d = *(const uint4*)(p + 260);
    A0.u[0]=a.x; A0.u[1]=a.y; A0.u[2]=a.z; A0.u[3]=a.w;
    A0.u[4]=b.x; A0.u[5]=b.y; A0.u[6]=b.z; A0.u[7]=b.w;
    A1.u[0]=c.x; A1.u[1]=c.y; A1.u[2]=c.z; A1.u[3]=c.w;
    A1.u[4]=d.x; A1.u[5]=d.y; A1.u[6]=d.z; A1.u[7]=d.w;
  }

  for (int ts = 0; ts < SEQT; ++ts) {
    // Prefetch next step's operands; latency hidden behind WMMAs + gates.
    lbase += ladv;
    if (ts + 1 < SEQT) {
      const uint32_t* p = h0u + lbase;
      uint4 a = *(const uint4*)(p);       uint4 b = *(const uint4*)(p + 4);
      uint4 c = *(const uint4*)(p + 256); uint4 d = *(const uint4*)(p + 260);
      N0.u[0]=a.x; N0.u[1]=a.y; N0.u[2]=a.z; N0.u[3]=a.w;
      N0.u[4]=b.x; N0.u[5]=b.y; N0.u[6]=b.z; N0.u[7]=b.w;
      N1.u[0]=c.x; N1.u[1]=c.y; N1.u[2]=c.z; N1.u[3]=c.w;
      N1.u[4]=d.x; N1.u[5]=d.y; N1.u[6]=d.z; N1.u[7]=d.w;
    }

    const v8f zero = {};
    v8f a0 = WMMA_F16(A0.v, bwi0[0], zero);
    v8f a1 = WMMA_F16(A0.v, bwi0[1], zero);
    v8f a2 = WMMA_F16(A0.v, bwi0[2], zero);
    v8f a3 = WMMA_F16(A0.v, bwi0[3], zero);
    a0 = WMMA_F16(A1.v, bwi1[0], a0);
    a1 = WMMA_F16(A1.v, bwi1[1], a1);
    a2 = WMMA_F16(A1.v, bwi1[2], a2);
    a3 = WMMA_F16(A1.v, bwi1[3], a3);
    a0 = WMMA_F16(Ah.v, bwh[0], a0);
    a1 = WMMA_F16(Ah.v, bwh[1], a1);
    a2 = WMMA_F16(Ah.v, bwh[2], a2);
    a3 = WMMA_F16(Ah.v, bwh[3], a3);
    v8f p4 = WMMA_F16(A0.v, bwi0[4], zero);
    v8f p5 = WMMA_F16(A0.v, bwi0[5], zero);
    p4 = WMMA_F16(A1.v, bwi1[4], p4);
    p5 = WMMA_F16(A1.v, bwi1[5], p5);
    v8f g4 = WMMA_F16(Ah.v, bwh[4], zero);
    v8f g5 = WMMA_F16(Ah.v, bwh[5], zero);

#pragma unroll
    for (int r = 0; r < 8; ++r) {
      const int m = r + 8 * hi;
      {
        const float rg = fsig(a0[r] + bc[0]);
        const float zg = fsig(a2[r] + bc[2]);
        const float ng = ftanh_(p4[r] + bin[0] + rg * (g4[r] + bhn[0]));
        const float hn = ng + zg * (hreg[r] - ng);
        hreg[r] = hn;
        pool[r] += hn;
        lds_h[m * HID + nloc] = (_Float16)hn;
      }
      {
        const float rg = fsig(a1[r] + bc[1]);
        const float zg = fsig(a3[r] + bc[3]);
        const float ng = ftanh_(p5[r] + bin[1] + rg * (g5[r] + bhn[1]));
        const float hn = ng + zg * (hreg[8 + r] - ng);
        hreg[8 + r] = hn;
        pool[8 + r] += hn;
        lds_h[m * HID + 16 + nloc] = (_Float16)hn;
      }
    }
    __syncthreads();

    // Rebuild recurrent A operand from the updated state.
    {
      const uint32_t* hrow = (const uint32_t*)&lds_h[nloc * HID];
#pragma unroll
      for (int i = 0; i < 4; ++i) { Ah.u[i] = hrow[hi * 8 + i]; Ah.u[4 + i] = hrow[hi * 8 + 4 + i]; }
    }
    __syncthreads();

    // Rotate prefetched operands in.
#pragma unroll
    for (int i = 0; i < 8; ++i) { A0.u[i] = N0.u[i]; A1.u[i] = N1.u[i]; }
  }

  const float inv = 1.f / (float)SEQT;
#pragma unroll
  for (int r = 0; r < 8; ++r) {
    const int m = r + 8 * hi;
    pooled[(size_t)(b0 + m) * 64 + dir * 32 + nloc]      = pool[r] * inv;
    pooled[(size_t)(b0 + m) * 64 + dir * 32 + 16 + nloc] = pool[8 + r] * inv;
  }
}

// ---------------------------------------------------------------------------
// Head: pooled[B,64] -> relu(fc1) -> fc2 -> out[B,1]. One thread per row.
// ---------------------------------------------------------------------------
__global__ void head_fc(const float* __restrict__ pooled,
                        const float* __restrict__ fc1w, const float* __restrict__ fc1b,
                        const float* __restrict__ fc2w, const float* __restrict__ fc2b,
                        float* __restrict__ out)
{
  const int b = blockIdx.x * blockDim.x + threadIdx.x;
  if (b >= BATCH) return;
  const float* p = pooled + (size_t)b * 64;
  float acc2 = fc2b[0];
#pragma unroll
  for (int i = 0; i < 16; ++i) {
    float a = fc1b[i];
#pragma unroll
    for (int j = 0; j < 64; ++j) a += p[j] * fc1w[i * 64 + j];
    acc2 += fmaxf(a, 0.f) * fc2w[i];
  }
  out[b] = acc2;
}

extern "C" void kernel_launch(void* const* d_in, const int* in_sizes, int n_in,
                              void* d_out, int out_size, void* d_ws, size_t ws_size,
                              hipStream_t stream) {
  (void)in_sizes; (void)n_in; (void)out_size; (void)ws_size;
  const float* x     = (const float*)d_in[0];
  const float* Wih0f = (const float*)d_in[1];
  const float* Whh0f = (const float*)d_in[2];
  const float* bih0f = (const float*)d_in[3];
  const float* bhh0f = (const float*)d_in[4];
  const float* Wih0b = (const float*)d_in[5];
  const float* Whh0b = (const float*)d_in[6];
  const float* bih0b = (const float*)d_in[7];
  const float* bhh0b = (const float*)d_in[8];
  const float* Wih1f = (const float*)d_in[9];
  const float* Whh1f = (const float*)d_in[10];
  const float* bih1f = (const float*)d_in[11];
  const float* bhh1f = (const float*)d_in[12];
  const float* Wih1b = (const float*)d_in[13];
  const float* Whh1b = (const float*)d_in[14];
  const float* bih1b = (const float*)d_in[15];
  const float* bhh1b = (const float*)d_in[16];
  const float* fc1w  = (const float*)d_in[17];
  const float* fc1b  = (const float*)d_in[18];
  const float* fc2w  = (const float*)d_in[19];
  const float* fc2b  = (const float*)d_in[20];

  _Float16* h0  = (_Float16*)d_ws;                                        // swizzled [tile][T][dir][lane][8dw]
  float* pooled = (float*)((char*)d_ws +
                           (size_t)BATCH * SEQT * 64 * sizeof(_Float16)); // [B][64] f32

  gru_layer0<<<dim3(64), dim3(32), 0, stream>>>(
      x, Wih0f, Whh0f, bih0f, bhh0f, Wih0b, Whh0b, bih0b, bhh0b, h0);
  gru_layer1<<<dim3(64), dim3(32), 0, stream>>>(
      h0, Wih1f, Whh1f, bih1f, bhh1f, Wih1b, Whh1b, bih1b, bhh1b, pooled);
  head_fc<<<dim3(2), dim3(256), 0, stream>>>(pooled, fc1w, fc1b, fc2w, fc2b, (float*)d_out);
}
